// ContinuousExponentialKoopmanOperator_52484500357778
// MI455X (gfx1250) — matrix operand order repaired and checked by reading, not compiled, not run on hardware
//
#include <hip/hip_runtime.h>

typedef __attribute__((ext_vector_type(16))) _Float16 v16h;
typedef __attribute__((ext_vector_type(8)))  _Float16 v8h;
typedef __attribute__((ext_vector_type(8)))  float    v8f;

#define DDIM    256
#define TSTEPS  1024
#define NCHUNK  8
#define CHUNKLEN 128   /* scan computes 127 steps; f32 seed owns slot 127 */
#define DT      0.01f

// ---------------- f32 helpers (expm Taylor + squaring chain) ----------------

// P = s*K + I
__global__ __launch_bounds__(256) void k_initP(const float* __restrict__ K,
                                               float* __restrict__ P, float s) {
  int id = blockIdx.x * 256 + threadIdx.x;      // 0..65535
  int r = id >> 8, c = id & 255;
  P[id] = s * K[id] + ((r == c) ? 1.0f : 0.0f);
}

// C = alpha*(A@B) + addI*I   (256x256, f32). 64 WGs: 16 row-tiles x 4 col-tiles.
__global__ __launch_bounds__(256) void k_mm256(const float* __restrict__ A,
                                               const float* __restrict__ B,
                                               float* __restrict__ C,
                                               float alpha, int addI) {
  __shared__ float sA[16 * DDIM];               // 16KB row strip of A
  int rt = blockIdx.x >> 2, ct = blockIdx.x & 3;
  int tid = threadIdx.x;
  for (int i = tid; i < 16 * DDIM; i += 256) {
    int m = i >> 8, k = i & 255;
    sA[i] = A[(rt * 16 + m) * DDIM + k];
  }
  __syncthreads();
  int n  = ct * 64 + (tid & 63);
  int mg = tid >> 6;                            // 4 row-groups of 4 rows
  float a0 = 0.f, a1 = 0.f, a2 = 0.f, a3 = 0.f;
  for (int k = 0; k < DDIM; ++k) {
    float b = B[k * DDIM + n];
    a0 += sA[(mg * 4 + 0) * DDIM + k] * b;
    a1 += sA[(mg * 4 + 1) * DDIM + k] * b;
    a2 += sA[(mg * 4 + 2) * DDIM + k] * b;
    a3 += sA[(mg * 4 + 3) * DDIM + k] * b;
  }
  float acc[4] = {a0, a1, a2, a3};
  for (int i = 0; i < 4; ++i) {
    int row = rt * 16 + mg * 4 + i;
    C[row * DDIM + n] = alpha * acc[i] + ((addI && row == n) ? 1.0f : 0.0f);
  }
}

// Dst_f16[n*256+k] = (f16) Src_f32[k*256+n]  (transpose so B-operand loads are contiguous)
__global__ __launch_bounds__(256) void k_transpose_cvt(const float* __restrict__ S,
                                                       _Float16* __restrict__ Dm) {
  int id = blockIdx.x * 256 + threadIdx.x;      // 0..65535
  int n = id >> 8, k = id & 255;
  Dm[id] = (_Float16)S[k * DDIM + n];           // id == n*256+k -> coalesced store
}

// ---------------- WMMA building blocks ----------------

__device__ __forceinline__ v16h cat16(v8h lo, v8h hi) {
  return __builtin_shufflevector(lo, hi, 0, 1, 2, 3, 4, 5, 6, 7,
                                         8, 9, 10, 11, 12, 13, 14, 15);
}

// out[rowbase.., t_out] = cvt_f16(in_rows) @ ET128   (one f16 WMMA matmul, f32 out)
// These outputs ARE re-read (chunk startup states), so keep regular-temporal stores.
__global__ __launch_bounds__(512) void k_seed(const float* __restrict__ in, long irs,
                                              float* __restrict__ outp, long ors,
                                              const _Float16* __restrict__ ETp) {
  __shared__ _Float16 zb[16 * DDIM];            // 8KB
  int tid = threadIdx.x;
  int rowbase = blockIdx.x * 16;
  for (int i = tid; i < 16 * DDIM; i += 512) {
    int m = i >> 8, k = i & 255;
    zb[i] = (_Float16)in[(long)(rowbase + m) * irs + k];
  }
  __syncthreads();
  int lane = tid & 31;
  int nn = lane & 15, hl = lane >> 4;
  int nbase = (tid >> 5) * 16;                  // wave's N-tile
  v8f acc = {};
#pragma unroll
  for (int c = 0; c < 8; ++c) {
    const v8h* bp = (const v8h*)(ETp + (nbase + nn) * DDIM + 32 * c + 16 * hl);
    v16h b = cat16(bp[0], bp[1]);
    v8h alo = *(const v8h*)(&zb[nn * DDIM + 32 * c + 16 * hl]);
    v8h ahi = *(const v8h*)(&zb[nn * DDIM + 32 * c + 16 * hl + 8]);
    v16h a = cat16(alo, ahi);
    acc = __builtin_amdgcn_wmma_f32_16x16x32_f16(false, a, false, b,
                                                 (short)0, acc, false, false);
  }
#pragma unroll
  for (int r = 0; r < 8; ++r)
    outp[(long)(rowbase + r + 8 * hl) * ors + nbase + nn] = acc[r];
}

// Main time-scan: grid (64 row-blocks x 8 chunks), 16 waves per WG.
// B (wave's 256x16 slice of E^T, f16) is register-resident across all steps.
// Output is 1 GiB of write-once streaming data (>> 192MB L2) -> non-temporal stores.
__global__ __launch_bounds__(512) void k_scan(const float* __restrict__ z0,
                                              const _Float16* __restrict__ ETp,
                                              float* __restrict__ outp) {
  __shared__ _Float16 zb[2][16 * DDIM];         // 16KB double-buffered state tile
  int tid = threadIdx.x;
  int rowbase = blockIdx.x * 16;
  int chunk = blockIdx.y;

  {   // chunk start state: z0 (chunk 0) or the f32 seed written at t = 128*chunk-1
    const float* src; long srs;
    if (chunk == 0) { src = z0; srs = DDIM; }
    else { src = outp + (long)(CHUNKLEN * chunk - 1) * DDIM; srs = (long)TSTEPS * DDIM; }
    for (int i = tid; i < 16 * DDIM; i += 512) {
      int m = i >> 8, k = i & 255;
      zb[0][i] = (_Float16)src[(long)(rowbase + m) * srs + k];
    }
  }

  int lane = tid & 31;
  int nn = lane & 15, hl = lane >> 4;
  int nbase = (tid >> 5) * 16;

  v16h B[8];                                    // 64 VGPRs, loaded once
#pragma unroll
  for (int c = 0; c < 8; ++c) {
    const v8h* bp = (const v8h*)(ETp + (nbase + nn) * DDIM + 32 * c + 16 * hl);
    B[c] = cat16(bp[0], bp[1]);
  }
  __syncthreads();

  float* op = outp + (long)(rowbase + 8 * hl) * TSTEPS * DDIM
                   + (long)(CHUNKLEN * chunk) * DDIM + nbase + nn;
  int sidx = nbase + nn;                        // LDS store column
  int am = nn;                                  // A-operand row (lane&15)

  for (int j = 0; j < CHUNKLEN - 1; ++j) {      // 127 steps; slot 127 is the f32 seed
    const _Float16* ib = zb[j & 1];
    _Float16* ob = zb[(j & 1) ^ 1];
    v8f acc = {};
#pragma unroll
    for (int c = 0; c < 8; ++c) {
      v8h alo = *(const v8h*)(&ib[am * DDIM + 32 * c + 16 * hl]);
      v8h ahi = *(const v8h*)(&ib[am * DDIM + 32 * c + 16 * hl + 8]);
      v16h a = cat16(alo, ahi);
      acc = __builtin_amdgcn_wmma_f32_16x16x32_f16(false, a, false, B[c],
                                                   (short)0, acc, false, false);
    }
#pragma unroll
    for (int r = 0; r < 8; ++r) {               // C layout: reg r -> row r+8*hl, lane -> col
      float v = acc[r];
      __builtin_nontemporal_store(v, &op[(long)r * TSTEPS * DDIM]);  // streaming f32 out
      ob[(r + 8 * hl) * DDIM + sidx] = (_Float16)v;   // f16 state for next step
    }
    op += DDIM;
    __syncthreads();
  }
}

// ---------------- host-side orchestration ----------------

extern "C" void kernel_launch(void* const* d_in, const int* in_sizes, int n_in,
                              void* d_out, int out_size, void* d_ws, size_t ws_size,
                              hipStream_t stream) {
  const float* z0 = (const float*)d_in[0];
  const float* K  = (const float*)d_in[1];
  float* out = (float*)d_out;
  char* ws = (char*)d_ws;

  float*    W0    = (float*)ws;                                    // 256 KB
  float*    W1    = (float*)(ws + (size_t)DDIM * DDIM * 4);        // 256 KB
  _Float16* ET    = (_Float16*)(ws + (size_t)2 * DDIM * DDIM * 4); // 128 KB  E^T  (f16)
  _Float16* ET128 = ET + (size_t)DDIM * DDIM;                      // 128 KB  (E^128)^T

  // E = expm(DT*K) via order-5 Horner Taylor, all f32 (||DT*K|| ~ 0.02 -> ~1e-13 trunc)
  k_initP<<<256, 256, 0, stream>>>(K, W0, DT / 5.0f);              // I + (DT/5)K
  k_mm256<<<64, 256, 0, stream>>>(K, W0, W1, DT / 4.0f, 1);
  k_mm256<<<64, 256, 0, stream>>>(K, W1, W0, DT / 3.0f, 1);
  k_mm256<<<64, 256, 0, stream>>>(K, W0, W1, DT / 2.0f, 1);
  k_mm256<<<64, 256, 0, stream>>>(K, W1, W0, DT, 1);               // W0 = E
  k_transpose_cvt<<<256, 256, 0, stream>>>(W0, ET);

  // E^128 by 7 f32 squarings
  k_mm256<<<64, 256, 0, stream>>>(W0, W0, W1, 1.0f, 0);            // E^2
  k_mm256<<<64, 256, 0, stream>>>(W1, W1, W0, 1.0f, 0);            // E^4
  k_mm256<<<64, 256, 0, stream>>>(W0, W0, W1, 1.0f, 0);            // E^8
  k_mm256<<<64, 256, 0, stream>>>(W1, W1, W0, 1.0f, 0);            // E^16
  k_mm256<<<64, 256, 0, stream>>>(W0, W0, W1, 1.0f, 0);            // E^32
  k_mm256<<<64, 256, 0, stream>>>(W1, W1, W0, 1.0f, 0);            // E^64
  k_mm256<<<64, 256, 0, stream>>>(W0, W0, W1, 1.0f, 0);            // E^128 -> W1
  k_transpose_cvt<<<256, 256, 0, stream>>>(W1, ET128);

  // Seed chunk boundaries: y_{128c-1} = y_{128(c-1)-1} @ E^128  (written into d_out)
  for (int c = 1; c <= NCHUNK; ++c) {
    const float* in = (c == 1) ? z0 : out + (long)(CHUNKLEN * (c - 1) - 1) * DDIM;
    long irs = (c == 1) ? (long)DDIM : (long)TSTEPS * DDIM;
    float* o = out + (long)(CHUNKLEN * c - 1) * DDIM;
    k_seed<<<64, 512, 0, stream>>>(in, irs, o, (long)TSTEPS * DDIM, ET128);
  }

  // 8 time-chunks x 64 row-blocks, each scans 127 steps with WMMA
  k_scan<<<dim3(64, NCHUNK), 512, 0, stream>>>(z0, ET, out);

  (void)in_sizes; (void)n_in; (void)out_size; (void)ws_size;
}
